// ScaledDotProductAttention_47425028882927
// MI455X (gfx1250) — matrix operand order repaired and checked by reading, not compiled, not run on hardware
//
#include <hip/hip_runtime.h>

typedef __attribute__((ext_vector_type(16))) _Float16 v16h;
typedef __attribute__((ext_vector_type(8)))  _Float16 v8h;
typedef __attribute__((ext_vector_type(8)))  float    v8f;

#define B_  32
#define N_  2048
#define M_  2048
#define D_  128

// 1/sqrt(128) (reference divides qk by temper before softmax)
#define INV_TEMPER 0.08838834764831845f

// ---------------------------------------------------------------------------
// Prep: fp32 -> fp16 conversion (vector-4)
// ---------------------------------------------------------------------------
__global__ __launch_bounds__(256) void cvt_f32_f16_kernel(
    const float* __restrict__ src, _Float16* __restrict__ dst, int n) {
  int i = (blockIdx.x * blockDim.x + threadIdx.x) * 4;
  if (i < n) {
    float4 f = *(const float4*)(src + i);
    union { _Float16 h[4]; uint2 u; } o;
    o.h[0] = (_Float16)f.x; o.h[1] = (_Float16)f.y;
    o.h[2] = (_Float16)f.z; o.h[3] = (_Float16)f.w;
    *(uint2*)(dst + i) = o.u;
  }
}

// ---------------------------------------------------------------------------
// Prep: V[b][m][d] (f32) -> Vt[b][d][m] (f16), 32x32 LDS tile transpose
// ---------------------------------------------------------------------------
__global__ __launch_bounds__(256) void transpose_v_kernel(
    const float* __restrict__ V, _Float16* __restrict__ Vt) {
  __shared__ _Float16 tile[32][33];
  const int b  = blockIdx.z;
  const int m0 = blockIdx.x * 32;
  const int d0 = blockIdx.y * 32;
  const int tx = threadIdx.x;        // 0..31
  const int ty = threadIdx.y;        // 0..7
  const float* src = V + ((size_t)b * M_ + m0) * D_ + d0;
#pragma unroll
  for (int j = 0; j < 32; j += 8)
    tile[ty + j][tx] = (_Float16)src[(size_t)(ty + j) * D_ + tx];
  __syncthreads();
  _Float16* dst = Vt + ((size_t)b * D_ + d0) * M_ + m0;
#pragma unroll
  for (int j = 0; j < 32; j += 8)
    dst[(size_t)(ty + j) * M_ + tx] = tile[tx][ty + j];
}

// ---------------------------------------------------------------------------
// Fragment loaders (wave32 WMMA layouts, ISA 7.12.2)
// ---------------------------------------------------------------------------
// A (16x32 f16): lane row = lane&15; lanes 0-15 K 0..15, lanes 16-31 K 16..31 (contiguous)
__device__ inline v16h load_a_frag(const _Float16* rowptr, int kbase, bool hi) {
  const _Float16* p = rowptr + kbase + (hi ? 16 : 0);
  v8h lo = *(const v8h*)(p);
  v8h hh = *(const v8h*)(p + 8);
  v16h a;
#pragma unroll
  for (int e = 0; e < 8; ++e) { a[e] = lo[e]; a[e + 8] = hh[e]; }
  return a;
}

// ---------------------------------------------------------------------------
// Pass A: per-column softmax stats over the n axis.
//   Cmax[b][m] = max_n (Q·K^T)[n,m] / temper
//   Rz[b][m]   = 1 / sum_n exp(s[n,m] - Cmax[b][m])
// One wave per 16 m-columns; streams all n in 16-row WMMA tiles. B-fragments
// (K columns) are cached in registers for the whole n loop.
// ---------------------------------------------------------------------------
__global__ __launch_bounds__(256) void attn_stats_kernel(
    const _Float16* __restrict__ Qh, const _Float16* __restrict__ Kh,
    float* __restrict__ Cmax, float* __restrict__ Rz) {
  const int b      = blockIdx.x;
  const int wave   = threadIdx.x >> 5;
  const int lane   = threadIdx.x & 31;
  const int lane16 = lane & 15;
  const bool hi    = lane >= 16;
  const int m0     = blockIdx.y * 128 + wave * 16;

  const _Float16* Qb = Qh + (size_t)b * N_ * D_;
  const _Float16* Kb = Kh + (size_t)b * M_ * D_;

  // Cache B-fragments: columns m0..m0+15, K-dim = d (4 blocks of 32)
  v16h bfrag[4];
  {
    const _Float16* krow = Kb + (size_t)(m0 + lane16) * D_;
#pragma unroll
    for (int kb = 0; kb < 4; ++kb)
      bfrag[kb] = *(const v16h*)(krow + kb * 32 + (hi ? 16 : 0));
  }

  float runm = -INFINITY, runz = 0.0f;
  for (int n0 = 0; n0 < N_; n0 += 16) {
    const _Float16* qrow = Qb + (size_t)(n0 + lane16) * D_;
    v8f s = {};
#pragma unroll
    for (int kb = 0; kb < 4; ++kb) {
      v16h a = load_a_frag(qrow, kb * 32, hi);
      s = __builtin_amdgcn_wmma_f32_16x16x32_f16(false, a, false, bfrag[kb],
                                                 (short)0, s, false, false);
    }
    // This lane owns column m = m0+lane16, rows i (+8 if hi) -> lane-local stats
    float sv[8], tmax = -INFINITY;
#pragma unroll
    for (int i = 0; i < 8; ++i) { sv[i] = s[i] * INV_TEMPER; tmax = fmaxf(tmax, sv[i]); }
    float newm = fmaxf(runm, tmax);
    float zi = 0.0f;
#pragma unroll
    for (int i = 0; i < 8; ++i) zi += __expf(sv[i] - newm);
    runz = runz * __expf(runm - newm) + zi;
    runm = newm;
  }

  // Merge row-halves: lane L holds rows 0..7 partials, lane L+16 rows 8..15
  float om = __shfl_xor(runm, 16, 32);
  float oz = __shfl_xor(runz, 16, 32);
  float cm = fmaxf(runm, om);
  float Z  = runz * __expf(runm - cm) + oz * __expf(om - cm);
  if (!hi) {
    size_t idx = (size_t)b * M_ + m0 + lane16;
    Cmax[idx] = cm;
    Rz[idx]   = 1.0f / Z;
  }
}

// ---------------------------------------------------------------------------
// Pass B: O[n-tile] = sum over m-blocks of softmax-weighted V.
// Wave owns 16 n-rows; loops m in blocks of 32:
//   S (16x32) via 8 WMMA  ->  W = exp(s-c[m])*rZ[m]  ->  LDS restage
//   O += W(16x32) x V(32x128) via 8 WMMA (B-frags contiguous from V^T)
// ---------------------------------------------------------------------------
__global__ __launch_bounds__(256) void attn_out_kernel(
    const _Float16* __restrict__ Qh, const _Float16* __restrict__ Kh,
    const _Float16* __restrict__ Vt, const float* __restrict__ Cmax,
    const float* __restrict__ Rz, float* __restrict__ out) {
  __shared__ __attribute__((aligned(32))) _Float16 Wt[8][16][32];

  const int b      = blockIdx.x;
  const int wave   = threadIdx.x >> 5;
  const int lane   = threadIdx.x & 31;
  const int lane16 = lane & 15;
  const bool hi    = lane >= 16;
  const int n0     = blockIdx.y * 128 + wave * 16;

  const _Float16* Qb = Qh + (size_t)b * N_ * D_;
  const _Float16* Kb = Kh + (size_t)b * M_ * D_;
  const _Float16* Vb = Vt + (size_t)b * D_ * M_;   // [d][m]
  const float*    Cb = Cmax + (size_t)b * M_;
  const float*    Rb = Rz   + (size_t)b * M_;

  // Cache this wave's Q A-fragments (16 rows x full D) for the whole m loop
  v16h afrag[4];
  {
    const _Float16* qrow = Qb + (size_t)(n0 + lane16) * D_;
#pragma unroll
    for (int kb = 0; kb < 4; ++kb) afrag[kb] = load_a_frag(qrow, kb * 32, hi);
  }

  v8f oacc[8];
#pragma unroll
  for (int dt = 0; dt < 8; ++dt) oacc[dt] = (v8f){};

  for (int m0 = 0; m0 < M_; m0 += 32) {
    // Prefetch next K block into caches (global_prefetch_b8)
    if (m0 + 32 < M_)
      __builtin_prefetch(Kb + (size_t)(m0 + 32 + lane16) * D_, 0, 0);

    // ---- S = Q K^T / temper, softmax weights, stage into LDS (16 x 32) ----
#pragma unroll
    for (int t = 0; t < 2; ++t) {
      const int mb = m0 + t * 16;
      const _Float16* krow = Kb + (size_t)(mb + lane16) * D_;
      v8f s = {};
#pragma unroll
      for (int kb = 0; kb < 4; ++kb) {
        v16h bf = *(const v16h*)(krow + kb * 32 + (hi ? 16 : 0));
        s = __builtin_amdgcn_wmma_f32_16x16x32_f16(false, afrag[kb], false, bf,
                                                   (short)0, s, false, false);
      }
      const float cm = Cb[mb + lane16];
      const float rz = Rb[mb + lane16];
#pragma unroll
      for (int i = 0; i < 8; ++i) {
        float w = __expf(s[i] * INV_TEMPER - cm) * rz;
        int row = hi ? (i + 8) : i;
        Wt[wave][row][t * 16 + lane16] = (_Float16)w;
      }
    }
    // Wave-local LDS RAW: drain DS counter before re-reading
    asm volatile("s_wait_dscnt 0x0" ::: "memory");

    // ---- A-fragment of W from LDS ----
    v16h wa;
    {
      const _Float16* wrow = &Wt[wave][lane16][hi ? 16 : 0];
      v8h lo = *(const v8h*)(wrow);
      v8h hh = *(const v8h*)(wrow + 8);
#pragma unroll
      for (int e = 0; e < 8; ++e) { wa[e] = lo[e]; wa[e + 8] = hh[e]; }
    }

    // ---- O += W x V : B-fragments contiguous rows of V^T ----
#pragma unroll
    for (int dt = 0; dt < 8; ++dt) {
      const _Float16* vp =
          Vb + (size_t)(dt * 16 + lane16) * M_ + m0 + (hi ? 16 : 0);
      v16h bf = *(const v16h*)vp;
      oacc[dt] = __builtin_amdgcn_wmma_f32_16x16x32_f16(false, wa, false, bf,
                                                        (short)0, oacc[dt],
                                                        false, false);
    }
  }

  // ---- store O (f32): lane owns column dt*16+lane16, rows i(+8 if hi) ----
  float* ob = out + ((size_t)b * N_ + n0) * D_;
#pragma unroll
  for (int dt = 0; dt < 8; ++dt) {
    const int col = dt * 16 + lane16;
#pragma unroll
    for (int i = 0; i < 8; ++i) {
      int row = hi ? (i + 8) : i;
      ob[(size_t)row * D_ + col] = oacc[dt][i];
    }
  }
}

// ---------------------------------------------------------------------------
// Host-side launcher
// ---------------------------------------------------------------------------
extern "C" void kernel_launch(void* const* d_in, const int* in_sizes, int n_in,
                              void* d_out, int out_size, void* d_ws, size_t ws_size,
                              hipStream_t stream) {
  (void)in_sizes; (void)n_in; (void)out_size; (void)ws_size;
  const float* Q = (const float*)d_in[0];
  const float* K = (const float*)d_in[1];
  const float* V = (const float*)d_in[2];
  float* out = (float*)d_out;

  // Workspace layout: Qh(16MB) | Kh(16MB) | Vt(16MB) | Cmax(256KB) | Rz(256KB)
  const size_t ELEMS = (size_t)B_ * N_ * D_;   // 8,388,608
  _Float16* Qh = (_Float16*)d_ws;
  _Float16* Kh = Qh + ELEMS;
  _Float16* Vt = Kh + ELEMS;
  float* Cmax  = (float*)(Vt + ELEMS);
  float* Rz    = Cmax + (size_t)B_ * M_;

  const int ncvt = (int)ELEMS;
  const int cvt_blocks = (ncvt / 4 + 255) / 256;
  cvt_f32_f16_kernel<<<cvt_blocks, 256, 0, stream>>>(Q, Qh, ncvt);
  cvt_f32_f16_kernel<<<cvt_blocks, 256, 0, stream>>>(K, Kh, ncvt);
  transpose_v_kernel<<<dim3(M_ / 32, D_ / 32, B_), dim3(32, 8), 0, stream>>>(V, Vt);

  attn_stats_kernel<<<dim3(B_, M_ / 128), 256, 0, stream>>>(Qh, Kh, Cmax, Rz);
  attn_out_kernel<<<dim3(B_, N_ / 128), 256, 0, stream>>>(Qh, Kh, Vt, Cmax, Rz, out);
}
